// RNNTDecoder_46883863003233
// MI455X (gfx1250) — compile-verified, matrix operand order checked by reading, not run on hardware
//
#include <hip/hip_runtime.h>
#include <hip/hip_bf16.h>
#include <math.h>

// ---------------------------------------------------------------------------
// RNN-T decoder for MI455X (gfx1250, wave32, WMMA).
// B=8, T=200, U1=101, V1=1025, ENC=PRED=512, EMB=128.
// ---------------------------------------------------------------------------

#define B_   8
#define T_   200
#define U1_  101
#define V1_  1025
#define P_   512
#define E_   128
#define G4_  2048   // 4*PRED

typedef __attribute__((ext_vector_type(16))) _Float16 v16h;
typedef __attribute__((ext_vector_type(8)))  _Float16 v8h;
typedef __attribute__((ext_vector_type(8)))  float    v8f;

static __device__ __forceinline__ int imin(int a, int b) { return a < b ? a : b; }
static __device__ __forceinline__ float sigmoidf_(float x) { return 1.0f / (1.0f + expf(-x)); }

static __device__ __forceinline__ v16h frag_from(const v8h lo, const v8h hi) {
    return __builtin_shufflevector(lo, hi, 0,1,2,3,4,5,6,7,8,9,10,11,12,13,14,15);
}

// ---------------------------------------------------------------------------
// f32 -> f16 convert
// ---------------------------------------------------------------------------
__global__ void k_cvt_f16(const float* __restrict__ src, _Float16* __restrict__ dst, int n) {
    int i = blockIdx.x * blockDim.x + threadIdx.x;
    if (i < n) dst[i] = (_Float16)src[i];
}

// ---------------------------------------------------------------------------
// Embedding gather -> f16   emb16[b][u][e], u=0 is blank (row VOCAB=1024)
// ---------------------------------------------------------------------------
__global__ void k_embed(const float* __restrict__ table, const int* __restrict__ targets,
                        _Float16* __restrict__ emb16) {
    int i = blockIdx.x * blockDim.x + threadIdx.x;
    int total = B_ * U1_ * E_;
    if (i >= total) return;
    int e = i & (E_ - 1);
    int u = (i / E_) % U1_;
    int b = i / (E_ * U1_);
    int tok = (u == 0) ? (V1_ - 1) : targets[b * (U1_ - 1) + (u - 1)];
    emb16[i] = (_Float16)table[(size_t)tok * E_ + e];
}

// ---------------------------------------------------------------------------
// Generic WMMA GEMM-NT:  out[MxN] f32 = X[MxK] f16 @ W[NxK]^T f16 + bias0 + bias1
// Block = 128 threads = 4 waves; each wave owns one 16x16 n-tile.
// grid.x = ceil(M/16), grid.y = ceil(N/64).
// ---------------------------------------------------------------------------
__global__ void k_gemm_nt(const _Float16* __restrict__ X, const _Float16* __restrict__ W,
                          const float* __restrict__ bias0, const float* __restrict__ bias1,
                          float* __restrict__ out, int M, int N, int K) {
    int wave = threadIdx.x >> 5;
    int lane = threadIdx.x & 31;
    int m0 = blockIdx.x * 16;
    int n0 = blockIdx.y * 64 + wave * 16;
    if (n0 >= N) return;

    int mrow = imin(m0 + (lane & 15), M - 1);
    int ncol = imin(n0 + (lane & 15), N - 1);
    int ksel = (lane >> 4) << 3;

    v8f c = {};
    for (int kb = 0; kb < K; kb += 32) {
        const v8h* pa = (const v8h*)(X + (size_t)mrow * K + kb + ksel);
        const v8h* pb = (const v8h*)(W + (size_t)ncol * K + kb + ksel);
        v16h a = frag_from(pa[0], pa[2]);   // chunk1 at +16 elements == +2 v8h
        v16h b = frag_from(pb[0], pb[2]);
        c = __builtin_amdgcn_wmma_f32_16x16x32_f16(false, a, false, b, (short)0, c, false, false);
    }

    float bv = (bias0 ? bias0[ncol] : 0.0f) + (bias1 ? bias1[ncol] : 0.0f);
    int n = n0 + (lane & 15);
    int mbase = m0 + ((lane >> 4) << 3);
#pragma unroll
    for (int r = 0; r < 8; ++r) {
        int m = mbase + r;
        if (m < M && n < N) out[(size_t)m * N + n] = c[r] + bv;
    }
}

// ---------------------------------------------------------------------------
// Recurrent LSTM: gates_x already holds  x@W_ih^T + b_ih + b_hh.
// One block per batch, 1024 threads (2 gate pre-activations each).
// ---------------------------------------------------------------------------
__global__ void k_lstm(const float* __restrict__ gates_x,      // (B,U1,2048)
                       const _Float16* __restrict__ Whh16,     // (2048,512)
                       _Float16* __restrict__ pred16) {        // (B,U1,512)
    int b = blockIdx.x;
    int tid = threadIdx.x;
    __shared__ float hsh[P_];
    __shared__ float csh[P_];
    __shared__ float gsh[G4_];

    if (tid < P_) { hsh[tid] = 0.0f; csh[tid] = 0.0f; }
    __syncthreads();

    int j0 = tid * 2, j1 = j0 + 1;
    const v8h* w0 = (const v8h*)(Whh16 + (size_t)j0 * P_);
    const v8h* w1 = (const v8h*)(Whh16 + (size_t)j1 * P_);

    for (int u = 0; u < U1_; ++u) {
        const float* gx = gates_x + ((size_t)b * U1_ + u) * G4_;
        float acc0 = gx[j0], acc1 = gx[j1];
        for (int t = 0; t < P_ / 8; ++t) {
            v8h a = w0[t];
            v8h bb = w1[t];
#pragma unroll
            for (int i = 0; i < 8; ++i) {
                float hv = hsh[t * 8 + i];
                acc0 += (float)a[i] * hv;
                acc1 += (float)bb[i] * hv;
            }
        }
        gsh[j0] = acc0; gsh[j1] = acc1;
        __syncthreads();
        if (tid < P_) {
            int p = tid;
            float ig = sigmoidf_(gsh[p]);
            float fg = sigmoidf_(gsh[P_ + p]);
            float gg = tanhf(gsh[2 * P_ + p]);
            float og = sigmoidf_(gsh[3 * P_ + p]);
            float cc = fg * csh[p] + ig * gg;
            float hh = og * tanhf(cc);
            csh[p] = cc; hsh[p] = hh;
            pred16[((size_t)b * U1_ + u) * P_ + p] = (_Float16)hh;
        }
        __syncthreads();
    }
}

// ---------------------------------------------------------------------------
// Joint: logits[bt][u][v] = tanh(f_enc[bt]+f_pred[b,u]) @ W_joint^T + b_joint
// grid = (1600, 2); block = 256 = 8 waves.
// Block computes a 64-row (u) x 512 (k) tanh tile once into 64 KB LDS; each
// wave then sweeps v-tiles, computing a 64x16 output slab per tile: one B
// fragment (L2) feeds FOUR wmma ops -> 4x less L2 traffic on W_joint.
// ---------------------------------------------------------------------------
#define MROWS 64
__global__ void k_joint(const float* __restrict__ f_enc,     // (1600,512)
                        const float* __restrict__ f_pred,    // (808,512)
                        const _Float16* __restrict__ Wj16,   // (1025,512)
                        const float* __restrict__ b_joint,   // (1025)
                        float* __restrict__ out) {           // (1600,101,1025)
    int bt = blockIdx.x;
    int b  = bt / T_;
    int u0 = blockIdx.y * MROWS;

    __shared__ __align__(16) _Float16 Asmem[MROWS * P_];   // 64 KB

    for (int idx = threadIdx.x; idx < MROWS * P_; idx += blockDim.x) {
        int r = idx >> 9;                 // 0..63
        int p = idx & (P_ - 1);
        int u = imin(u0 + r, U1_ - 1);    // clamp padded rows (stores masked)
        float v = tanhf(f_enc[(size_t)bt * P_ + p] + f_pred[((size_t)b * U1_ + u) * P_ + p]);
        Asmem[idx] = (_Float16)v;
    }
    __syncthreads();

    int wave = threadIdx.x >> 5;
    int lane = threadIdx.x & 31;
    int arow = lane & 15;
    int ksel = (lane >> 4) << 3;
    const int NVT = (V1_ + 15) / 16;   // 65 v-tiles

    for (int vt = wave; vt < NVT; vt += 8) {
        int n0 = vt * 16;
        int ncol = imin(n0 + (lane & 15), V1_ - 1);
        {   // pull next v-tile's W rows toward L2/L0
            int ncn = imin(n0 + 128 + (lane & 15), V1_ - 1);
            __builtin_prefetch(Wj16 + (size_t)ncn * P_, 0, 1);
        }
        v8f c0 = {}, c1 = {}, c2 = {}, c3 = {};
        for (int kb = 0; kb < P_; kb += 32) {
            const v8h* pb = (const v8h*)(Wj16 + (size_t)ncol * P_ + kb + ksel);
            v16h bfrag = frag_from(pb[0], pb[2]);

            const v8h* pa0 = (const v8h*)(&Asmem[(arow     ) * P_ + kb + ksel]);
            const v8h* pa1 = (const v8h*)(&Asmem[(arow + 16) * P_ + kb + ksel]);
            const v8h* pa2 = (const v8h*)(&Asmem[(arow + 32) * P_ + kb + ksel]);
            const v8h* pa3 = (const v8h*)(&Asmem[(arow + 48) * P_ + kb + ksel]);
            v16h a0 = frag_from(pa0[0], pa0[2]);
            v16h a1 = frag_from(pa1[0], pa1[2]);
            v16h a2 = frag_from(pa2[0], pa2[2]);
            v16h a3 = frag_from(pa3[0], pa3[2]);

            c0 = __builtin_amdgcn_wmma_f32_16x16x32_f16(false, a0, false, bfrag, (short)0, c0, false, false);
            c1 = __builtin_amdgcn_wmma_f32_16x16x32_f16(false, a1, false, bfrag, (short)0, c1, false, false);
            c2 = __builtin_amdgcn_wmma_f32_16x16x32_f16(false, a2, false, bfrag, (short)0, c2, false, false);
            c3 = __builtin_amdgcn_wmma_f32_16x16x32_f16(false, a3, false, bfrag, (short)0, c3, false, false);
        }
        float bv = b_joint[ncol];
        int n = n0 + (lane & 15);
        bool nok = (n < V1_);
        int mb = ((lane >> 4) << 3);
#pragma unroll
        for (int r = 0; r < 8; ++r) {
            int m = mb + r;
            int u = u0 + m;
            size_t base = ((size_t)bt * U1_) * V1_ + n;
            if (nok && (u      < U1_)) out[base + (size_t)(u     ) * V1_] = c0[r] + bv;
            if (nok && (u + 16 < U1_)) out[base + (size_t)(u + 16) * V1_] = c1[r] + bv;
            if (nok && (u + 32 < U1_)) out[base + (size_t)(u + 32) * V1_] = c2[r] + bv;
            if (nok && (u + 48 < U1_)) out[base + (size_t)(u + 48) * V1_] = c3[r] + bv;
        }
    }
}

// ---------------------------------------------------------------------------
// Host-side orchestration
// ---------------------------------------------------------------------------
extern "C" void kernel_launch(void* const* d_in, const int* in_sizes, int n_in,
                              void* d_out, int out_size, void* d_ws, size_t ws_size,
                              hipStream_t stream) {
    const float* enc_out  = (const float*)d_in[0];   // (8,200,512)
    const int*   targets  = (const int*)  d_in[1];   // (8,100)
    const float* emb_tab  = (const float*)d_in[3];   // (1025,128)
    const float* W_ih     = (const float*)d_in[4];   // (2048,128)
    const float* W_hh     = (const float*)d_in[5];   // (2048,512)
    const float* b_ih     = (const float*)d_in[6];
    const float* b_hh     = (const float*)d_in[7];
    const float* W_enc    = (const float*)d_in[8];   // (512,512)
    const float* b_enc    = (const float*)d_in[9];
    const float* W_pred   = (const float*)d_in[10];  // (512,512)
    const float* b_pred   = (const float*)d_in[11];
    const float* W_joint  = (const float*)d_in[12];  // (1025,512)
    const float* b_joint  = (const float*)d_in[13];
    float* out = (float*)d_out;

    char* cur = (char*)d_ws;
    auto carve = [&](size_t bytes) -> void* {
        void* p = (void*)cur;
        cur += (bytes + 255) & ~(size_t)255;
        return p;
    };
    _Float16* emb16   = (_Float16*)carve((size_t)B_ * U1_ * E_ * 2);
    _Float16* enc16   = (_Float16*)carve((size_t)B_ * T_ * P_ * 2);
    _Float16* Wih16   = (_Float16*)carve((size_t)G4_ * E_ * 2);
    _Float16* Whh16   = (_Float16*)carve((size_t)G4_ * P_ * 2);
    _Float16* Wenc16  = (_Float16*)carve((size_t)P_ * P_ * 2);
    _Float16* Wpred16 = (_Float16*)carve((size_t)P_ * P_ * 2);
    _Float16* Wj16    = (_Float16*)carve((size_t)V1_ * P_ * 2);
    float*    gates_x = (float*)   carve((size_t)B_ * U1_ * G4_ * 4);
    _Float16* pred16  = (_Float16*)carve((size_t)B_ * U1_ * P_ * 2);
    float*    f_enc   = (float*)   carve((size_t)B_ * T_ * P_ * 4);
    float*    f_pred  = (float*)   carve((size_t)B_ * U1_ * P_ * 4);

    auto cvt = [&](const float* s, _Float16* d, int n) {
        k_cvt_f16<<<(n + 255) / 256, 256, 0, stream>>>(s, d, n);
    };
    cvt(W_ih,    Wih16,   G4_ * E_);
    cvt(W_hh,    Whh16,   G4_ * P_);
    cvt(W_enc,   Wenc16,  P_ * P_);
    cvt(W_pred,  Wpred16, P_ * P_);
    cvt(W_joint, Wj16,    V1_ * P_);
    cvt(enc_out, enc16,   B_ * T_ * P_);

    {   // embedding gather
        int n = B_ * U1_ * E_;
        k_embed<<<(n + 255) / 256, 256, 0, stream>>>(emb_tab, targets, emb16);
    }

    // gates_x = emb @ W_ih^T + b_ih + b_hh      (M=808, N=2048, K=128)
    {
        int M = B_ * U1_, N = G4_, K = E_;
        dim3 g((M + 15) / 16, (N + 63) / 64);
        k_gemm_nt<<<g, 128, 0, stream>>>(emb16, Wih16, b_ih, b_hh, gates_x, M, N, K);
    }

    // recurrent LSTM (sequential over U1 steps), one block per batch
    k_lstm<<<B_, 1024, 0, stream>>>(gates_x, Whh16, pred16);

    // f_enc = enc @ W_enc^T + b_enc             (M=1600, N=512, K=512)
    {
        int M = B_ * T_, N = P_, K = P_;
        dim3 g((M + 15) / 16, (N + 63) / 64);
        k_gemm_nt<<<g, 128, 0, stream>>>(enc16, Wenc16, b_enc, nullptr, f_enc, M, N, K);
    }

    // f_pred = pred @ W_pred^T + b_pred         (M=808, N=512, K=512)
    {
        int M = B_ * U1_, N = P_, K = P_;
        dim3 g((M + 15) / 16, (N + 63) / 64);
        k_gemm_nt<<<g, 128, 0, stream>>>(pred16, Wpred16, b_pred, nullptr, f_pred, M, N, K);
    }

    // joint: tanh(f_enc + f_pred) @ W_joint^T + b_joint  -> logits
    {
        dim3 g(B_ * T_, 2);
        k_joint<<<g, 256, 0, stream>>>(f_enc, f_pred, Wj16, b_joint, out);
    }
}
